// transformer_17609365914203
// MI455X (gfx1250) — matrix operand order repaired and checked by reading, not compiled, not run on hardware
//
#include <hip/hip_runtime.h>
#include <hip/hip_bf16.h>
#include <stdint.h>

#define B_    4
#define N_    8192
#define CIN   128
#define M_    2048
#define K_    20
#define HEADS_ 8
#define DIMK_ 32
#define COUT_ 256
#define DIMV_ 32
#define CR    128       // CH_RAISE
#define CEFF  288       // 2*C_IN + DIM_V
#define LTOT  (B_*M_*K_)  // 163840
#define EPS_  1e-5f

typedef __attribute__((ext_vector_type(16))) __bf16 v16bf;
typedef __attribute__((ext_vector_type(8)))  __bf16 v8bf;
typedef __attribute__((ext_vector_type(8)))  float  v8f;

static __device__ __forceinline__ unsigned short f2bf(float f) {
  unsigned u = __float_as_uint(f);
  u += 0x7FFFu + ((u >> 16) & 1u);           // round-to-nearest-even
  return (unsigned short)(u >> 16);
}
static __device__ __forceinline__ float bf2f(unsigned short h) {
  return __uint_as_float(((unsigned)h) << 16);
}

// ---------------------------------------------------------------------------
// 1) Furthest point sampling: one 1024-thread block per batch.
//    Each thread owns 8 points: coords + running min-dist live in REGISTERS,
//    so the 2048-step serial loop touches no global memory for distances.
// ---------------------------------------------------------------------------
__global__ __launch_bounds__(1024)
void fps_kernel(const float* __restrict__ xyz, const float* __restrict__ feat,
                int* __restrict__ fps_idx, float* __restrict__ new_xyz,
                float* __restrict__ new_feat)
{
  __shared__ float wbv[32];
  __shared__ int   wbi[32];
  __shared__ float cpt[3];
  __shared__ int   sfar;
  const int b = blockIdx.x;
  const int tid = threadIdx.x;
  const int lane = tid & 31, wv = tid >> 5;
  const float* bx = xyz  + (size_t)b * N_ * 3;
  const float* bf = feat + (size_t)b * N_ * CIN;

  float px[8], py[8], pz[8], pd[8];
  #pragma unroll
  for (int s = 0; s < 8; ++s) {
    const int n = tid + s * 1024;
    px[s] = bx[n * 3 + 0]; py[s] = bx[n * 3 + 1]; pz[s] = bx[n * 3 + 2];
    pd[s] = 1e10f;
  }
  if (tid == 0) sfar = 0;
  __syncthreads();

  for (int m = 0; m < M_; ++m) {
    const int far = sfar;
    if (tid == 0) fps_idx[b * M_ + m] = far;
    if ((far & 1023) == tid) {                 // owner broadcasts centroid
      const int s = far >> 10;
      float ox = px[0], oy = py[0], oz = pz[0];
      #pragma unroll
      for (int t = 1; t < 8; ++t) if (s == t) { ox = px[t]; oy = py[t]; oz = pz[t]; }
      cpt[0] = ox; cpt[1] = oy; cpt[2] = oz;
      new_xyz[((size_t)b * M_ + m) * 3 + 0] = ox;
      new_xyz[((size_t)b * M_ + m) * 3 + 1] = oy;
      new_xyz[((size_t)b * M_ + m) * 3 + 2] = oz;
    }
    if (tid < CIN) new_feat[((size_t)b * M_ + m) * CIN + tid] = bf[(size_t)far * CIN + tid];
    __syncthreads();

    const float c0 = cpt[0], c1 = cpt[1], c2 = cpt[2];
    float bvv = -1.0f; int bii = 0;
    #pragma unroll
    for (int s = 0; s < 8; ++s) {
      float dx = px[s] - c0, dy = py[s] - c1, dz = pz[s] - c2;
      float d  = dx * dx + dy * dy + dz * dz;
      float dn = fminf(pd[s], d);
      pd[s] = dn;
      if (dn > bvv) { bvv = dn; bii = tid + s * 1024; }
    }
    // wave32 argmax reduce, lowest-index tiebreak
    #pragma unroll
    for (int off = 16; off > 0; off >>= 1) {
      float ov = __shfl_xor(bvv, off, 32);
      int   oi = __shfl_xor(bii, off, 32);
      if (ov > bvv || (ov == bvv && oi < bii)) { bvv = ov; bii = oi; }
    }
    if (lane == 0) { wbv[wv] = bvv; wbi[wv] = bii; }
    __syncthreads();
    if (wv == 0) {
      float v2 = wbv[lane]; int i2 = wbi[lane];
      #pragma unroll
      for (int off = 16; off > 0; off >>= 1) {
        float ov = __shfl_xor(v2, off, 32);
        int   oi = __shfl_xor(i2, off, 32);
        if (ov > v2 || (ov == v2 && oi < i2)) { v2 = ov; i2 = oi; }
      }
      if (lane == 0) sfar = i2;
    }
    __syncthreads();
  }
}

// ---------------------------------------------------------------------------
// 2) KNN: one wave per query, 8 queries per block (block-aligned to batch).
//    xyz staged through LDS in 1024-point chunks shared by all 8 waves.
//    Per-lane sorted top-20 lists in LDS, merged via shuffle-min + ballot.
// ---------------------------------------------------------------------------
#define CHUNK_ 1024
__global__ __launch_bounds__(256)
void knn_kernel(const float* __restrict__ xyz, const float* __restrict__ new_xyz,
                int* __restrict__ knn_idx)
{
  __shared__ float sx[CHUNK_ * 3];           // 12 KB
  __shared__ float kd[8 * 32 * K_];          // 20 KB
  __shared__ int   ki[8 * 32 * K_];          // 20 KB
  const int tid = threadIdx.x, lane = tid & 31, wv = tid >> 5;
  const int q = blockIdx.x * 8 + wv;         // all 8 queries share batch b
  const int b = q / M_;
  const float* bx = xyz + (size_t)b * N_ * 3;
  const float qx = new_xyz[(size_t)q * 3 + 0];
  const float qy = new_xyz[(size_t)q * 3 + 1];
  const float qz = new_xyz[(size_t)q * 3 + 2];
  const int base = (wv * 32 + lane) * K_;

  for (int j = 0; j < K_; ++j) { kd[base + j] = 3.0e38f; ki[base + j] = 0; }

  for (int c0 = 0; c0 < N_; c0 += CHUNK_) {
    __syncthreads();
    for (int i = tid; i < CHUNK_ * 3; i += 256) sx[i] = bx[(size_t)c0 * 3 + i];
    __syncthreads();
    for (int n = lane; n < CHUNK_; n += 32) {
      float dx = sx[n * 3 + 0] - qx, dy = sx[n * 3 + 1] - qy, dz = sx[n * 3 + 2] - qz;
      float d = dx * dx + dy * dy + dz * dz;
      if (d < kd[base + K_ - 1]) {
        int j = K_ - 1;
        while (j > 0 && kd[base + j - 1] > d) {
          kd[base + j] = kd[base + j - 1]; ki[base + j] = ki[base + j - 1]; --j;
        }
        kd[base + j] = d; ki[base + j] = c0 + n;
      }
    }
  }
  // merge 32 sorted lists -> global top-20
  int p = 0;
  for (int j = 0; j < K_; ++j) {
    float myv = (p < K_) ? kd[base + p] : 3.0e38f;
    float mv = myv;
    #pragma unroll
    for (int off = 16; off > 0; off >>= 1) mv = fminf(mv, __shfl_xor(mv, off, 32));
    unsigned long long bal = __ballot(myv == mv);
    int winner = __ffsll(bal) - 1;
    if (lane == winner) { knn_idx[(size_t)q * K_ + j] = ki[base + p]; ++p; }
  }
}

// ---------------------------------------------------------------------------
// 3) Build the 288-channel grouped feature, column-major bf16
//    (288 contiguous channels per (b,m,k) column) -> direct WMMA B-operand.
// ---------------------------------------------------------------------------
__global__ __launch_bounds__(288)
void build_feat_kernel(const float* __restrict__ xyz, const float* __restrict__ feat,
                       const float* __restrict__ new_xyz, const float* __restrict__ new_feat,
                       const int* __restrict__ knn_idx, const float* __restrict__ Wh,
                       unsigned short* __restrict__ featB)
{
  __shared__ float srel[3];
  __shared__ int   sidx;
  const int l = blockIdx.x;          // column in [0, LTOT)
  const int c = threadIdx.x;         // channel in [0, 288)
  const int p = l / K_;              // b*M + m
  const int b = p / M_;
  if (c == 0) sidx = knn_idx[l];
  __syncthreads();
  const int idx = sidx;
  if (c < 3) srel[c] = xyz[((size_t)b * N_ + idx) * 3 + c] - new_xyz[(size_t)p * 3 + c];
  __syncthreads();
  float val;
  if (c < CIN) {
    val = feat[((size_t)b * N_ + idx) * CIN + c] - new_feat[(size_t)p * CIN + c];
  } else if (c < 2 * CIN) {
    val = new_feat[(size_t)p * CIN + (c - CIN)];
  } else {
    const int v = c - 2 * CIN;
    val = Wh[v * 3 + 0] * srel[0] + Wh[v * 3 + 1] * srel[1] + Wh[v * 3 + 2] * srel[2];
  }
  featB[(size_t)l * CEFF + c] = f2bf(val);
}

// ---------------------------------------------------------------------------
// 4) Generic bf16 WMMA GEMM:  C[rowsA x L] = A[rowsA x KD] * B[KD x L]
//    Compile-time KD (288 or 128) and HAS_BNB so the K-loop is straight-line,
//    fully unrolled (9 or 4 v_wmma_f32_16x16x32_bf16 with clause'd b128 loads).
//    HAS_BNB: per-K-channel scale/bias + ReLU applied to B on the fly
//             (fuses previous layer's batch-norm; no f32 round-trip).
//    sums != null: per-out-channel sum/sum^2 (LDS ds_add_f32 -> 1 global
//                  atomic per channel per block) for batch-norm stats.
//    Cf/Cb: f32 and/or raw-bf16 output (packed uint4 stores).
//    A layout: lane(lrow,hi) holds row lrow, K = hi*16+{0..15}.
//    B layout (same k order as A): lane(lrow,hi) holds column lrow, K = hi*16+{0..15}.
// ---------------------------------------------------------------------------
template<int KD, bool HAS_BNB>
__global__ __launch_bounds__(512)
void gemm_bf16_kernel(const unsigned short* __restrict__ A,
                      const unsigned short* __restrict__ Bm,
                      float* __restrict__ Cf, unsigned short* __restrict__ Cb,
                      int rowsA,
                      const float* __restrict__ bnb, float* __restrict__ sums)
{
  __shared__ float ssum[256], ssq[256];
  const int tid  = threadIdx.x, lane = tid & 31, wv = tid >> 5;
  const int lrow = lane & 15,   hi   = lane >> 4;
  const int mo   = wv * 16;                 // out-row tile base for this wave
  const int col0 = blockIdx.x * 16;         // column tile base for this block
  if (sums) {
    for (int i = tid; i < rowsA; i += blockDim.x) { ssum[i] = 0.f; ssq[i] = 0.f; }
    __syncthreads();
  }

  v8f acc = {};
  const unsigned short* arow = A  + (size_t)(mo + lrow) * KD + hi * 16;
  const unsigned short* bcol = Bm + (size_t)(col0 + lrow) * KD + hi * 16;
  #pragma unroll
  for (int kb = 0; kb < KD; kb += 32) {
    v8bf a0 = *reinterpret_cast<const v8bf*>(arow + kb);
    v8bf a1 = *reinterpret_cast<const v8bf*>(arow + kb + 8);
    v16bf av;
    #pragma unroll
    for (int i = 0; i < 8; ++i) { av[i] = a0[i]; av[i + 8] = a1[i]; }
    v16bf bvec = *reinterpret_cast<const v16bf*>(bcol + kb);
    if (HAS_BNB) {                               // fused BN+ReLU on B operand
      const float* sc = bnb + kb + hi * 16;
      const float* bi = bnb + KD + kb + hi * 16;
      #pragma unroll
      for (int i = 0; i < 16; ++i) {
        float x = fmaxf(sc[i] * (float)bvec[i] + bi[i], 0.f);
        bvec[i] = (__bf16)x;
      }
    }
    acc = __builtin_amdgcn_wmma_f32_16x16x32_bf16(false, av, false, bvec,
                                                  (short)0, acc, false, false);
  }

  // D layout: VGPR r, lane(lrow,hi) -> row = mo + r + 8*hi, col = col0 + lrow
  if (Cf) {
    float* cc = Cf + (size_t)(col0 + lrow) * rowsA + mo + hi * 8;
    #pragma unroll
    for (int r = 0; r < 8; ++r) cc[r] = acc[r];
  }
  if (Cb) {
    unsigned short* cb = Cb + (size_t)(col0 + lrow) * rowsA + mo + hi * 8;
    uint4 o;
    o.x = (unsigned)f2bf(acc[0]) | ((unsigned)f2bf(acc[1]) << 16);
    o.y = (unsigned)f2bf(acc[2]) | ((unsigned)f2bf(acc[3]) << 16);
    o.z = (unsigned)f2bf(acc[4]) | ((unsigned)f2bf(acc[5]) << 16);
    o.w = (unsigned)f2bf(acc[6]) | ((unsigned)f2bf(acc[7]) << 16);
    *reinterpret_cast<uint4*>(cb) = o;
  }

  if (sums) {
    #pragma unroll
    for (int r = 0; r < 8; ++r) {
      const int row = mo + hi * 8 + r;
      atomicAdd(&ssum[row], acc[r]);
      atomicAdd(&ssq[row],  acc[r] * acc[r]);
    }
    __syncthreads();
    for (int i = tid; i < rowsA; i += blockDim.x) {
      atomicAdd(&sums[i], ssum[i]);
      atomicAdd(&sums[rowsA + i], ssq[i]);
    }
  }
}

// ---------------------------------------------------------------------------
// 5) BN finalize: per-channel scale/bias from (sum, sumsq).
// ---------------------------------------------------------------------------
__global__ void bn_finalize_kernel(const float* __restrict__ sums, int C, float invcnt,
                                   const float* __restrict__ gamma,
                                   const float* __restrict__ beta,
                                   float* __restrict__ sb)
{
  const int c = threadIdx.x + blockIdx.x * blockDim.x;
  if (c >= C) return;
  const float mean = sums[c] * invcnt;
  const float var  = sums[C + c] * invcnt - mean * mean;
  const float inv  = rsqrtf(var + EPS_);
  const float sc   = gamma[c] * inv;
  sb[c]     = sc;
  sb[C + c] = beta[c] - mean * sc;
}

// 6) BN + ReLU + max over K neighbors (reads raw bf16 h2) -> fq (f32 + bf16)
__global__ __launch_bounds__(128)
void apply_bn_relu_max_kernel(const unsigned short* __restrict__ H2b,
                              const float* __restrict__ sb,
                              float* __restrict__ fq, unsigned short* __restrict__ fqb)
{
  const int c = threadIdx.x;
  const int p = blockIdx.x;
  const float sc = sb[c], bi = sb[128 + c];
  float best = -3.0e38f;
  #pragma unroll
  for (int k = 0; k < K_; ++k)
    best = fmaxf(best, sc * bf2f(H2b[((size_t)p * K_ + k) * CR + c]) + bi);
  best = fmaxf(best, 0.f);   // relu(max) == max(relu) (monotone)
  fq [(size_t)p * CR + c] = best;
  fqb[(size_t)p * CR + c] = f2bf(best);
}

// ---------------------------------------------------------------------------
// 7) Per-point attention: softmax(K-logits) over K, BN(v), kv = kk*v^T,
//    out = q*kv. One 128-thread block per sampled point.
// ---------------------------------------------------------------------------
__global__ __launch_bounds__(128)
void attention_kernel(const float* __restrict__ qraw, const float* __restrict__ sbq,
                      const float* __restrict__ vraw, const float* __restrict__ sbv,
                      const float* __restrict__ kraw, float* __restrict__ out)
{
  __shared__ float kk[DIMK_][K_];
  __shared__ float vv[DIMV_][K_];
  __shared__ float kv[DIMK_][DIMV_];
  __shared__ float qv[COUT_];
  const int tid = threadIdx.x;
  const int p = blockIdx.x;
  const int b = p / M_, m = p % M_;

  if (tid < DIMK_) {
    const int d = tid;
    float tmp[K_];
    float mx = -3.0e38f;
    #pragma unroll
    for (int k = 0; k < K_; ++k) {
      tmp[k] = kraw[((size_t)p * K_ + k) * DIMK_ + d];
      mx = fmaxf(mx, tmp[k]);
    }
    float s = 0.f;
    #pragma unroll
    for (int k = 0; k < K_; ++k) { tmp[k] = __expf(tmp[k] - mx); s += tmp[k]; }
    const float inv = 1.f / s;
    #pragma unroll
    for (int k = 0; k < K_; ++k) kk[d][k] = tmp[k] * inv;
  }
  for (int i = tid; i < DIMV_ * K_; i += 128) {
    const int v = i / K_, k = i % K_;
    vv[v][k] = sbv[v] * vraw[((size_t)p * K_ + k) * DIMV_ + v] + sbv[DIMV_ + v];
  }
  for (int c = tid; c < COUT_; c += 128)
    qv[c] = fmaxf(sbq[c] * qraw[(size_t)p * COUT_ + c] + sbq[COUT_ + c], 0.f);
  __syncthreads();

  for (int o = tid; o < DIMK_ * DIMV_; o += 128) {
    const int d = o / DIMV_, v = o % DIMV_;
    float a = 0.f;
    #pragma unroll
    for (int k = 0; k < K_; ++k) a += kk[d][k] * vv[v][k];
    kv[d][v] = a;
  }
  __syncthreads();
  for (int o = tid; o < COUT_; o += 128) {
    const int h = o / DIMV_, v = o % DIMV_;
    float a = 0.f;
    #pragma unroll
    for (int d = 0; d < DIMK_; ++d) a += qv[h * DIMK_ + d] * kv[d][v];
    out[((size_t)b * COUT_ + o) * M_ + m] = a;
  }
}

// Small helpers
__global__ void zero_kernel(float* p, int n) {
  const int i = threadIdx.x + blockIdx.x * blockDim.x;
  if (i < n) p[i] = 0.f;
}
__global__ void cvt_kernel(const float* __restrict__ src, unsigned short* __restrict__ dst, int n) {
  const int i = threadIdx.x + blockIdx.x * blockDim.x;
  if (i < n) dst[i] = f2bf(src[i]);
}

// ---------------------------------------------------------------------------
extern "C" void kernel_launch(void* const* d_in, const int* in_sizes, int n_in,
                              void* d_out, int out_size, void* d_ws, size_t ws_size,
                              hipStream_t stream)
{
  (void)in_sizes; (void)n_in; (void)out_size; (void)ws_size;
  const float* xyz     = (const float*)d_in[0];
  const float* feature = (const float*)d_in[1];
  const float* W1 = (const float*)d_in[2];
  const float* g1 = (const float*)d_in[3];
  const float* b1 = (const float*)d_in[4];
  const float* W2 = (const float*)d_in[5];
  const float* g2 = (const float*)d_in[6];
  const float* b2 = (const float*)d_in[7];
  const float* Wv = (const float*)d_in[8];
  const float* gv = (const float*)d_in[9];
  const float* bv = (const float*)d_in[10];
  const float* Wk = (const float*)d_in[11];
  const float* Wq = (const float*)d_in[12];
  const float* gq = (const float*)d_in[13];
  const float* bq = (const float*)d_in[14];
  const float* Wh = (const float*)d_in[15];

  float* new_xyz = (float*)d_out;                       // [B, M, 3]
  float* out     = (float*)d_out + (size_t)B_ * M_ * 3; // [B, 256, M]

  char* w = (char*)d_ws;
  auto alloc = [&](size_t bytes) { char* r = w; w += (bytes + 255) & ~(size_t)255; return r; };
  int*            fps_idx = (int*)           alloc(sizeof(int)   * B_ * M_);
  float*          newfeat = (float*)         alloc(sizeof(float) * (size_t)B_ * M_ * CIN);
  int*            knn_idx = (int*)           alloc(sizeof(int)   * (size_t)LTOT);
  unsigned short* featB   = (unsigned short*)alloc(2ull * LTOT * CEFF);
  unsigned short* h1b     = (unsigned short*)alloc(2ull * LTOT * CR);   // raw (pre-BN)
  unsigned short* h2b     = (unsigned short*)alloc(2ull * LTOT * CR);   // raw (pre-BN)
  float*          fq      = (float*)         alloc(4ull * B_ * M_ * CR);
  unsigned short* fqb     = (unsigned short*)alloc(2ull * B_ * M_ * CR);
  float*          qraw    = (float*)         alloc(4ull * B_ * M_ * COUT_);
  float*          vraw    = (float*)         alloc(4ull * LTOT * DIMV_);
  float*          kraw    = (float*)         alloc(4ull * LTOT * DIMK_);
  unsigned short* W1b     = (unsigned short*)alloc(2ull * CR * CEFF);
  unsigned short* W2b     = (unsigned short*)alloc(2ull * CR * CR);
  unsigned short* Wvb     = (unsigned short*)alloc(2ull * DIMV_ * CEFF);
  unsigned short* Wkb     = (unsigned short*)alloc(2ull * DIMK_ * CEFF);
  unsigned short* Wqb     = (unsigned short*)alloc(2ull * COUT_ * CR);
  const int nsums = 2 * CR + 2 * CR + 2 * DIMV_ + 2 * COUT_;
  float* sums = (float*)alloc(4ull * nsums);
  float* sum1 = sums;
  float* sum2 = sums + 2 * CR;
  float* sumv = sums + 4 * CR;
  float* sumq = sums + 4 * CR + 2 * DIMV_;
  float* sb1  = (float*)alloc(4ull * 2 * CR);
  float* sb2  = (float*)alloc(4ull * 2 * CR);
  float* sbv  = (float*)alloc(4ull * 2 * DIMV_);
  float* sbq  = (float*)alloc(4ull * 2 * COUT_);

  zero_kernel<<<(nsums + 255) / 256, 256, 0, stream>>>(sums, nsums);
  cvt_kernel<<<(CR * CEFF + 255) / 256, 256, 0, stream>>>(W1, W1b, CR * CEFF);
  cvt_kernel<<<(CR * CR + 255) / 256, 256, 0, stream>>>(W2, W2b, CR * CR);
  cvt_kernel<<<(DIMV_ * CEFF + 255) / 256, 256, 0, stream>>>(Wv, Wvb, DIMV_ * CEFF);
  cvt_kernel<<<(DIMK_ * CEFF + 255) / 256, 256, 0, stream>>>(Wk, Wkb, DIMK_ * CEFF);
  cvt_kernel<<<(COUT_ * CR + 255) / 256, 256, 0, stream>>>(Wq, Wqb, COUT_ * CR);

  fps_kernel<<<B_, 1024, 0, stream>>>(xyz, feature, fps_idx, new_xyz, newfeat);
  knn_kernel<<<B_ * M_ / 8, 256, 0, stream>>>(xyz, new_xyz, knn_idx);
  build_feat_kernel<<<LTOT, CEFF, 0, stream>>>(xyz, feature, new_xyz, newfeat,
                                               knn_idx, Wh, featB);

  // MLP1: 128x288 @ 288xL -> raw bf16 h1 (+BN1 stats from f32 accumulators)
  gemm_bf16_kernel<CEFF, false><<<LTOT / 16, 256, 0, stream>>>(
      W1b, featB, nullptr, h1b, CR, nullptr, sum1);
  bn_finalize_kernel<<<1, CR, 0, stream>>>(sum1, CR, 1.0f / (float)LTOT, g1, b1, sb1);
  // MLP2: 128x128 @ 128xL, BN1+ReLU fused into B-operand load (+BN2 stats)
  gemm_bf16_kernel<CR, true><<<LTOT / 16, 256, 0, stream>>>(
      W2b, h1b, nullptr, h2b, CR, sb1, sum2);
  bn_finalize_kernel<<<1, CR, 0, stream>>>(sum2, CR, 1.0f / (float)LTOT, g2, b2, sb2);
  apply_bn_relu_max_kernel<<<B_ * M_, 128, 0, stream>>>(h2b, sb2, fq, fqb);
  // Q: 256x128 @ 128x(B*M)
  gemm_bf16_kernel<CR, false><<<B_ * M_ / 16, 512, 0, stream>>>(
      Wqb, fqb, qraw, nullptr, COUT_, nullptr, sumq);
  bn_finalize_kernel<<<1, COUT_, 0, stream>>>(sumq, COUT_, 1.0f / (float)(B_ * M_),
                                              gq, bq, sbq);
  // V: 32x288 @ 288xL (+BN stats);  K: 32x288 @ 288xL (softmax later, no BN)
  gemm_bf16_kernel<CEFF, false><<<LTOT / 16, 64, 0, stream>>>(
      Wvb, featB, vraw, nullptr, DIMV_, nullptr, sumv);
  bn_finalize_kernel<<<1, DIMV_, 0, stream>>>(sumv, DIMV_, 1.0f / (float)LTOT, gv, bv, sbv);
  gemm_bf16_kernel<CEFF, false><<<LTOT / 16, 64, 0, stream>>>(
      Wkb, featB, kraw, nullptr, DIMK_, nullptr, nullptr);

  attention_kernel<<<B_ * M_, 128, 0, stream>>>(qraw, sbq, vraw, sbv, kraw, out);
}
